// TorusEGNN_86131274154934
// MI455X (gfx1250) — matrix operand order repaired and checked by reading, not compiled, not run on hardware
//
#include <hip/hip_runtime.h>
#include <hip/hip_bf16.h>
#include <hip/hip_fp16.h>

#define B_   2
#define N_   512
#define D_   64
#define C_   3
#define H_   128
#define NT_I (N_/16)
#define JSPLIT 8
#define TWO_PI_F 6.283185307179586f
#define INV_2PI_F 0.15915494309189535f

typedef __attribute__((ext_vector_type(16))) _Float16 v16h;
typedef __attribute__((ext_vector_type(8)))  float    v8f;

union FragH { v16h v; unsigned int u[8]; uint4 q[2]; };
union H2    { _Float16 h[2]; unsigned int u; };

__device__ __forceinline__ float frcp(float x){ return __builtin_amdgcn_rcpf(x); }
__device__ __forceinline__ float siluf(float x){ return x * frcp(1.0f + __expf(-x)); }
__device__ __forceinline__ float sigmf(float x){ return frcp(1.0f + __expf(-x)); }
// A-matrix 16x32 f16 layout: element e of lane (m = L%16, g = L/16): K = e + 8g + (e>=8 ? 8 : 0)
__device__ __forceinline__ int akof(int kt, int e, int g){ return kt*32 + e + 8*g + ((e>=8)?8:0); }

__device__ __forceinline__ FragH ldfrag(const unsigned int* base, int tile, int lane){
  FragH f;
  const uint4* p = (const uint4*)(base + ((size_t)(tile*32 + lane))*8);
  f.q[0] = p[0]; f.q[1] = p[1];
  return f;
}

#define WMMA_F16(a,b,c) __builtin_amdgcn_wmma_f32_16x16x32_f16(false,(a),false,(b),(short)0,(c),false,false)
#define V8F_ZERO {0.f,0.f,0.f,0.f,0.f,0.f,0.f,0.f}

// ---- workspace layout (dwords) ----
#define OFF_WE2 0
#define SZ_WE2 (2*8*32*8)
#define OFF_WC1 (OFF_WE2+SZ_WE2)
#define SZ_WC1 (4*8*32*8)
#define OFF_WC2 (OFF_WC1+SZ_WC1)
#define SZ_WC2 (4*4*32*8)
#define SZ_EDGE_FRAGS (SZ_WE2+SZ_WC1+SZ_WC2)   /* 16384 dwords = 64KB, contiguous */
#define OFF_WN1 (OFF_WC2+SZ_WC2)
#define SZ_WN1 (6*8*32*8)
#define OFF_WN2 (OFF_WN1+SZ_WN1)
#define SZ_WN2 (4*4*32*8)
#define OFF_WN3 (OFF_WN2+SZ_WN2)
#define SZ_WN3 (2*4*32*8)
#define OFF_AV  (OFF_WN3+SZ_WN3)
#define SZ_AV   (B_*N_*64)
#define OFF_BV  (OFF_AV+SZ_AV)
#define SZ_BV   (B_*N_*64)
#define OFF_MI  (OFF_BV+SZ_BV)
#define SZ_MI   (B_*N_*H_)

// ---- format a (K x Nc) row-major f32 weight into WMMA B-fragments (f16) ----
// B-matrix 32x16 f16 layout: element e of lane (n = nt*16 + L%16, g = L/16): K = kt*32 + e + 8g + (e>=8 ? 8 : 0)
__global__ void prep_frags(const float* __restrict__ W, int K, int Nc, unsigned int* __restrict__ out){
  int t = blockIdx.x*blockDim.x + threadIdx.x;
  int KT = K>>5, NTc = Nc>>4;
  int total = KT*NTc*32;
  if (t >= total) return;
  int lane = t & 31, tile = t >> 5;
  int nt = tile % NTc, kt = tile / NTc;
  int g = lane >> 4, n = nt*16 + (lane & 15);
  unsigned int* dst = out + (size_t)t*8;
  for (int v=0; v<8; ++v){
    int K0 = kt*32 + 2*v + 8*g + ((v>=4) ? 8 : 0);
    H2 p; p.h[0] = (_Float16)W[(size_t)K0*Nc + n]; p.h[1] = (_Float16)W[(size_t)(K0+1)*Nc + n];
    dst[v] = p.u;
  }
}

// ---- per-node precompute: Av = feats @ We1[0:64] + b_e1 ; Bv = feats @ We1[64:128] ----
__global__ void prep_nodepre(const float* __restrict__ feats, const float* __restrict__ We1,
                             const float* __restrict__ be1,
                             float* __restrict__ Av, float* __restrict__ Bv){
  int t = blockIdx.x*blockDim.x + threadIdx.x;
  if (t >= B_*N_*64) return;
  int k = t & 63, row = t >> 6;
  const float* f = feats + (size_t)row*64;
  float a = be1[k], b = 0.f;
  for (int d=0; d<64; ++d){ a += f[d]*We1[d*64+k]; b += f[d]*We1[(64+d)*64+k]; }
  Av[t] = a; Bv[t] = b;
}

// ---- init accumulators: m_i = 0, coors_out = coors ----
__global__ void init_acc(float* __restrict__ Mi, float* __restrict__ outc, const float* __restrict__ coors){
  int t = blockIdx.x*blockDim.x + threadIdx.x;
  if (t < SZ_MI) Mi[t] = 0.f;
  if (t < B_*N_*C_) outc[t] = coors[t];
}

// ---- fused edge kernel ----
__global__ __launch_bounds__(128, 1) void edge_kernel(
    const float* __restrict__ coors, const float* __restrict__ Av, const float* __restrict__ Bv,
    const float* __restrict__ We1,
    const unsigned int* __restrict__ fWall,   // contiguous We2|Wc1|Wc2 fragments
    const float* __restrict__ be2, const float* __restrict__ Wg, const float* __restrict__ bg,
    const float* __restrict__ bc1, const float* __restrict__ bc2,
    const float* __restrict__ Wc3, const float* __restrict__ bc3,
    float* __restrict__ Mi, float* __restrict__ outc)
{
  // dynamic LDS: 16384 dwords of weight fragments (We2 | Wc1 | Wc2)
  extern __shared__ unsigned int dynWF[];
  __shared__ _Float16 ldsAll[4][16*128];
  __shared__ float    ldsGate[4][16];

  // cooperative one-time stage of all edge-MLP weight fragments into LDS
  {
    const uint4* src = (const uint4*)fWall;
    uint4* dst = (uint4*)dynWF;
    for (int t = threadIdx.x; t < SZ_EDGE_FRAGS/4; t += 128) dst[t] = src[t];
  }
  __syncthreads();
  const unsigned int* sWe2 = dynWF;
  const unsigned int* sWc1 = dynWF + SZ_WE2;
  const unsigned int* sWc2 = dynWF + SZ_WE2 + SZ_WC1;

  const int lane = threadIdx.x & 31, w = threadIdx.x >> 5;
  const int n0 = lane & 15, g = lane >> 4;
  _Float16* Lw = ldsAll[w];
  float*    Lg = ldsGate[w];

  const int bid = blockIdx.x;
  const int b   = bid / (NT_I*JSPLIT);
  const int rem = bid % (NT_I*JSPLIT);
  const int it  = rem / JSPLIT, js = rem % JSPLIT;
  const int i0  = it*16;
  const int jb  = js*(N_/JSPLIT) + w*16;

  const float* wr = We1 + 128*64;        // rel_dist row of W_e1
  const int i_own = i0 + n0;

  // hoisted per-lane invariants
  float avr[32], wrr[32];
  #pragma unroll
  for (int kt=0; kt<2; ++kt)
    #pragma unroll
    for (int e=0; e<16; ++e){
      int K = akof(kt,e,g);
      avr[kt*16+e] = Av[(size_t)(b*N_+i_own)*64 + K];
      wrr[kt*16+e] = wr[K];
    }
  float be2r[8], wgr[8], bc1r[8], bc2r[4], wc3r[4];
  #pragma unroll
  for (int nt=0; nt<8; ++nt){ int n=nt*16+n0; be2r[nt]=be2[n]; wgr[nt]=Wg[n]; bc1r[nt]=bc1[n]; }
  #pragma unroll
  for (int nt=0; nt<4; ++nt){ int n=nt*16+n0; bc2r[nt]=bc2[n]; wc3r[nt]=Wc3[n]; }
  const float bg0 = bg[0], bc30 = bc3[0];
  float cio[3];
  #pragma unroll
  for (int c=0;c<3;++c) cio[c] = coors[(size_t)(b*N_+i_own)*3 + c];

  // m_i accumulated in A-layout: this lane owns row i0+n0, columns akof(kt,e,g)
  float mi_acc[64];
  #pragma unroll
  for (int q=0; q<64; ++q) mi_acc[q] = 0.f;
  float cacc[8];
  #pragma unroll
  for (int r=0; r<8; ++r) cacc[r] = 0.f;

  #pragma unroll 1
  for (int jj=0; jj<16; ++jj){
    const int j = jb + jj;
    float cj[3];
    #pragma unroll
    for (int c=0;c<3;++c) cj[c] = coors[(size_t)(b*N_+j)*3 + c];

    // rel_dist for this lane's own row
    float rd = 0.f;
    #pragma unroll
    for (int c=0;c<3;++c){
      float rel = cio[c] - cj[c];
      rel -= TWO_PI_F * rintf(rel*INV_2PI_F);
      rd += rel*rel;
    }

    // build A-fragments of h1 = silu(A_i + B_j + rd*w_r): phase-separated for TRANS batching
    FragH a1[2];
    {
      const float* bvp = Bv + (size_t)(b*N_+j)*64;
      float x[32], sg[32];
      #pragma unroll
      for (int kt=0; kt<2; ++kt)
        #pragma unroll
        for (int e=0; e<16; ++e){
          int q = kt*16 + e;
          x[q] = avr[q] + bvp[akof(kt,e,g)] + rd*wrr[q];
        }
      #pragma unroll
      for (int q=0; q<32; ++q) sg[q] = __expf(-x[q]);
      #pragma unroll
      for (int q=0; q<32; ++q) sg[q] = frcp(1.0f + sg[q]);
      #pragma unroll
      for (int kt=0; kt<2; ++kt)
        #pragma unroll
        for (int v=0; v<8; ++v){
          int q = kt*16 + 2*v;
          H2 p;
          p.h[0] = (_Float16)(x[q]   * sg[q]);
          p.h[1] = (_Float16)(x[q+1] * sg[q+1]);
          a1[kt].u[v] = p.u;
        }
    }

    // GEMM1: [16x64] @ We2[64x128] -> ungated m to LDS, gate partials (double-buffered B-frags)
    float gp[8];
    #pragma unroll
    for (int r=0;r<8;++r) gp[r]=0.f;
    {
      FragH bcur0 = ldfrag(sWe2, 0*8+0, lane);
      FragH bcur1 = ldfrag(sWe2, 1*8+0, lane);
      #pragma unroll
      for (int nt=0; nt<8; ++nt){
        FragH bnxt0 = bcur0, bnxt1 = bcur1;
        if (nt < 7){
          bnxt0 = ldfrag(sWe2, 0*8+nt+1, lane);
          bnxt1 = ldfrag(sWe2, 1*8+nt+1, lane);
        }
        v8f acc = V8F_ZERO;
        acc = WMMA_F16(a1[0].v, bcur0.v, acc);
        acc = WMMA_F16(a1[1].v, bcur1.v, acc);
        float x[8], sg[8];
        #pragma unroll
        for (int r=0;r<8;++r) x[r] = acc[r] + be2r[nt];
        #pragma unroll
        for (int r=0;r<8;++r) sg[r] = __expf(-x[r]);
        #pragma unroll
        for (int r=0;r<8;++r) sg[r] = frcp(1.0f + sg[r]);
        #pragma unroll
        for (int r=0;r<8;++r){
          float m = x[r]*sg[r];
          gp[r] += m * wgr[nt];
          Lw[(r+8*g)*128 + nt*16 + n0] = (_Float16)m;   // ungated
        }
        bcur0 = bnxt0; bcur1 = bnxt1;
      }
    }
    // gate = sigmoid(m @ W_g + b_g): reduce over n within 16-lane group
    float gate[8];
    #pragma unroll
    for (int r=0;r<8;++r){
      float s = gp[r];
      s += __shfl_xor(s,1,16); s += __shfl_xor(s,2,16);
      s += __shfl_xor(s,4,16); s += __shfl_xor(s,8,16);
      gate[r] = sigmf(s + bg0);
    }
    // broadcast gate to A-layout consumers (gate indexed by row m)
    if (n0 == 0){
      #pragma unroll
      for (int r=0;r<8;++r) Lg[r + 8*g] = gate[r];
    }
    const float gm = Lg[n0];   // gate for row m = n0 (this lane's A-frag row)

    // gated A-fragments of m_ij + A-layout m_i accumulation
    FragH a2[4];
    #pragma unroll
    for (int kt=0; kt<4; ++kt)
      #pragma unroll
      for (int v=0; v<8; ++v){
        int K0 = akof(kt, 2*v, g);
        H2 hv; hv.u = *(const unsigned int*)&Lw[n0*128 + K0];
        float f0 = (float)hv.h[0] * gm;
        float f1 = (float)hv.h[1] * gm;
        mi_acc[kt*16 + 2*v]     += f0;
        mi_acc[kt*16 + 2*v + 1] += f1;
        H2 gv; gv.h[0] = (_Float16)f0; gv.h[1] = (_Float16)f1;
        a2[kt].u[v] = gv.u;
      }
    // GEMM2: [16x128] @ Wc1[128x128] -> silu -> LDS (double-buffered B-frags)
    {
      FragH bcur[4];
      #pragma unroll
      for (int kt=0; kt<4; ++kt) bcur[kt] = ldfrag(sWc1, kt*8+0, lane);
      #pragma unroll
      for (int nt=0; nt<8; ++nt){
        FragH bnxt[4] = {bcur[0], bcur[1], bcur[2], bcur[3]};
        if (nt < 7){
          #pragma unroll
          for (int kt=0; kt<4; ++kt) bnxt[kt] = ldfrag(sWc1, kt*8+nt+1, lane);
        }
        v8f acc = V8F_ZERO;
        #pragma unroll
        for (int kt=0; kt<4; ++kt) acc = WMMA_F16(a2[kt].v, bcur[kt].v, acc);
        float x[8], sg[8];
        #pragma unroll
        for (int r=0;r<8;++r) x[r] = acc[r] + bc1r[nt];
        #pragma unroll
        for (int r=0;r<8;++r) sg[r] = __expf(-x[r]);
        #pragma unroll
        for (int r=0;r<8;++r) sg[r] = frcp(1.0f + sg[r]);
        #pragma unroll
        for (int r=0;r<8;++r)
          Lw[(r+8*g)*128 + nt*16 + n0] = (_Float16)(x[r]*sg[r]);
        #pragma unroll
        for (int kt=0; kt<4; ++kt) bcur[kt] = bnxt[kt];
      }
    }
    // GEMM3: [16x128] @ Wc2[128x64] -> silu -> dot with Wc3 (double-buffered B-frags)
    FragH a3[4];
    #pragma unroll
    for (int kt=0; kt<4; ++kt)
      #pragma unroll
      for (int v=0; v<8; ++v){
        int K0 = akof(kt, 2*v, g);
        a3[kt].u[v] = *(const unsigned int*)&Lw[n0*128 + K0];
      }
    float cp[8];
    #pragma unroll
    for (int r=0;r<8;++r) cp[r]=0.f;
    {
      FragH bcur[4];
      #pragma unroll
      for (int kt=0; kt<4; ++kt) bcur[kt] = ldfrag(sWc2, kt*4+0, lane);
      #pragma unroll
      for (int nt=0; nt<4; ++nt){
        FragH bnxt[4] = {bcur[0], bcur[1], bcur[2], bcur[3]};
        if (nt < 3){
          #pragma unroll
          for (int kt=0; kt<4; ++kt) bnxt[kt] = ldfrag(sWc2, kt*4+nt+1, lane);
        }
        v8f acc = V8F_ZERO;
        #pragma unroll
        for (int kt=0; kt<4; ++kt) acc = WMMA_F16(a3[kt].v, bcur[kt].v, acc);
        float x[8], sg[8];
        #pragma unroll
        for (int r=0;r<8;++r) x[r] = acc[r] + bc2r[nt];
        #pragma unroll
        for (int r=0;r<8;++r) sg[r] = __expf(-x[r]);
        #pragma unroll
        for (int r=0;r<8;++r) sg[r] = frcp(1.0f + sg[r]);
        #pragma unroll
        for (int r=0;r<8;++r) cp[r] += (x[r]*sg[r]) * wc3r[nt];
        #pragma unroll
        for (int kt=0; kt<4; ++kt) bcur[kt] = bnxt[kt];
      }
    }
    float coorw[8];
    #pragma unroll
    for (int r=0;r<8;++r){
      float s = cp[r];
      s += __shfl_xor(s,1,16); s += __shfl_xor(s,2,16);
      s += __shfl_xor(s,4,16); s += __shfl_xor(s,8,16);
      coorw[r] = s + bc30;
    }
    // coordinate accumulation: lanes with n0 < 3 own coordinate c = n0
    if (n0 < 3){
      #pragma unroll
      for (int r=0;r<8;++r){
        int i = i0 + r + 8*g;
        float s2 = 0.f, relc = 0.f;
        #pragma unroll
        for (int c=0;c<3;++c){
          float rel = coors[(size_t)(b*N_+i)*3 + c] - cj[c];
          rel -= TWO_PI_F * rintf(rel*INV_2PI_F);
          s2 += rel*rel;
          if (c == n0) relc = rel;
        }
        cacc[r] += coorw[r] * relc * frcp(s2 + TWO_PI_F);
      }
    }
  }

  // write back partial accumulations (m_i in A-layout: row i0+n0)
  #pragma unroll
  for (int kt=0; kt<4; ++kt)
    #pragma unroll
    for (int e=0; e<16; ++e)
      unsafeAtomicAdd(&Mi[(size_t)(b*N_ + i0 + n0)*H_ + akof(kt,e,g)], mi_acc[kt*16+e]);
  if (n0 < 3){
    #pragma unroll
    for (int r=0;r<8;++r)
      unsafeAtomicAdd(&outc[(size_t)(b*N_ + i0 + r + 8*g)*3 + n0], cacc[r]);
  }
}

// ---- node MLP kernel: one wave per 16-row tile ----
__global__ __launch_bounds__(32) void node_kernel(
    const float* __restrict__ feats, const float* __restrict__ Mi,
    const unsigned int* __restrict__ fWn1, const unsigned int* __restrict__ fWn2, const unsigned int* __restrict__ fWn3,
    const float* __restrict__ bn1, const float* __restrict__ bn2, const float* __restrict__ bn3,
    float* __restrict__ outn)
{
  __shared__ _Float16 Lw[16*128];
  const int lane = threadIdx.x & 31;
  const int n0 = lane & 15, g = lane >> 4;
  const int bid = blockIdx.x;
  const int b = bid / NT_I, it = bid % NT_I;
  const int i0 = it*16;
  const int row = b*N_ + i0 + n0;

  // A fragments of x = [feats | m_i] (K = 192)
  FragH a[6];
  #pragma unroll
  for (int kt=0; kt<6; ++kt)
    #pragma unroll
    for (int v=0; v<8; ++v){
      int e0 = 2*v, e1 = 2*v+1;
      int K0 = akof(kt,e0,g), K1 = akof(kt,e1,g);
      float x0 = (K0 < 64) ? feats[(size_t)row*64 + K0] : Mi[(size_t)row*H_ + K0 - 64];
      float x1 = (K1 < 64) ? feats[(size_t)row*64 + K1] : Mi[(size_t)row*H_ + K1 - 64];
      H2 p; p.h[0] = (_Float16)x0; p.h[1] = (_Float16)x1;
      a[kt].u[v] = p.u;
    }
  // GEMM: [16x192] @ Wn1[192x128] -> silu -> LDS
  #pragma unroll
  for (int nt=0; nt<8; ++nt){
    v8f acc = V8F_ZERO;
    #pragma unroll
    for (int kt=0; kt<6; ++kt){
      FragH bf = ldfrag(fWn1, kt*8+nt, lane);
      acc = WMMA_F16(a[kt].v, bf.v, acc);
    }
    int n = nt*16 + n0;
    float bb = bn1[n];
    #pragma unroll
    for (int r=0;r<8;++r)
      Lw[(r+8*g)*128 + n] = (_Float16)siluf(acc[r] + bb);
  }
  // GEMM: [16x128] @ Wn2[128x64] -> silu -> LDS (stride 64)
  FragH a2[4];
  #pragma unroll
  for (int kt=0; kt<4; ++kt)
    #pragma unroll
    for (int v=0; v<8; ++v){
      int K0 = akof(kt, 2*v, g);
      a2[kt].u[v] = *(const unsigned int*)&Lw[n0*128 + K0];
    }
  #pragma unroll
  for (int nt=0; nt<4; ++nt){
    v8f acc = V8F_ZERO;
    #pragma unroll
    for (int kt=0; kt<4; ++kt){
      FragH bf = ldfrag(fWn2, kt*4+nt, lane);
      acc = WMMA_F16(a2[kt].v, bf.v, acc);
    }
    int n = nt*16 + n0;
    float bb = bn2[n];
    #pragma unroll
    for (int r=0;r<8;++r)
      Lw[(r+8*g)*64 + n] = (_Float16)siluf(acc[r] + bb);
  }
  // GEMM: [16x64] @ Wn3[64x64] + b + feats -> node_out
  FragH a3[2];
  #pragma unroll
  for (int kt=0; kt<2; ++kt)
    #pragma unroll
    for (int v=0; v<8; ++v){
      int K0 = akof(kt, 2*v, g);
      a3[kt].u[v] = *(const unsigned int*)&Lw[n0*64 + K0];
    }
  #pragma unroll
  for (int nt=0; nt<4; ++nt){
    v8f acc = V8F_ZERO;
    #pragma unroll
    for (int kt=0; kt<2; ++kt){
      FragH bf = ldfrag(fWn3, kt*4+nt, lane);
      acc = WMMA_F16(a3[kt].v, bf.v, acc);
    }
    int n = nt*16 + n0;
    float bb = bn3[n];
    #pragma unroll
    for (int r=0;r<8;++r){
      int gi = b*N_ + i0 + r + 8*g;
      outn[(size_t)gi*64 + n] = acc[r] + bb + feats[(size_t)gi*64 + n];
    }
  }
}

extern "C" void kernel_launch(void* const* d_in, const int* in_sizes, int n_in,
                              void* d_out, int out_size, void* d_ws, size_t ws_size,
                              hipStream_t stream) {
  (void)in_sizes; (void)n_in; (void)out_size; (void)ws_size;
  const float* feats = (const float*)d_in[0];
  const float* coors = (const float*)d_in[1];
  const float* We1   = (const float*)d_in[2];
  const float* be1   = (const float*)d_in[3];
  const float* We2   = (const float*)d_in[4];
  const float* be2   = (const float*)d_in[5];
  const float* Wg    = (const float*)d_in[6];
  const float* bg    = (const float*)d_in[7];
  const float* Wc1   = (const float*)d_in[8];
  const float* bc1   = (const float*)d_in[9];
  const float* Wc2   = (const float*)d_in[10];
  const float* bc2   = (const float*)d_in[11];
  const float* Wc3   = (const float*)d_in[12];
  const float* bc3   = (const float*)d_in[13];
  const float* Wn1   = (const float*)d_in[14];
  const float* bn1   = (const float*)d_in[15];
  const float* Wn2   = (const float*)d_in[16];
  const float* bn2   = (const float*)d_in[17];
  const float* Wn3   = (const float*)d_in[18];
  const float* bn3   = (const float*)d_in[19];

  float*        ws  = (float*)d_ws;
  unsigned int* wsu = (unsigned int*)d_ws;
  float* Av   = ws + OFF_AV;
  float* Bv   = ws + OFF_BV;
  float* Mi   = ws + OFF_MI;
  float* outn = (float*)d_out;
  float* outc = (float*)d_out + (size_t)B_*N_*D_;

  init_acc<<<(SZ_MI+255)/256, 256, 0, stream>>>(Mi, outc, coors);
  prep_frags<<<2, 256, 0, stream>>>(We2,  64, 128, wsu + OFF_WE2);
  prep_frags<<<4, 256, 0, stream>>>(Wc1, 128, 128, wsu + OFF_WC1);
  prep_frags<<<2, 256, 0, stream>>>(Wc2, 128,  64, wsu + OFF_WC2);
  prep_frags<<<6, 256, 0, stream>>>(Wn1, 192, 128, wsu + OFF_WN1);
  prep_frags<<<2, 256, 0, stream>>>(Wn2, 128,  64, wsu + OFF_WN2);
  prep_frags<<<1, 256, 0, stream>>>(Wn3,  64,  64, wsu + OFF_WN3);
  prep_nodepre<<<(B_*N_*64+127)/128, 128, 0, stream>>>(feats, We1, be1, Av, Bv);

  edge_kernel<<<B_*NT_I*JSPLIT, 128, (size_t)SZ_EDGE_FRAGS*4, stream>>>(
      coors, Av, Bv, We1,
      wsu + OFF_WE2,
      be2, Wg, bg, bc1, bc2, Wc3, bc3, Mi, outc);

  node_kernel<<<B_*NT_I, 32, 0, stream>>>(
      feats, Mi, wsu + OFF_WN1, wsu + OFF_WN2, wsu + OFF_WN3,
      bn1, bn2, bn3, outn);
}
